// LoopClosurePNA_87565793231059
// MI455X (gfx1250) — compile-verified, matrix operand order checked by reading, not run on hardware
//
#include <hip/hip_runtime.h>
#include <hip/hip_bf16.h>
#include <stdint.h>

#define N_NODES   100000
#define N_EDGES   800000
#define NODE_F    64
#define HID       32
#define TOWERS    4
#define F_OUT_T   8
#define OUT_F     8
#define NUM_GRAPHS 64
#define AVG_DEG_LOG 2.19722457733621938f   /* log(9) */
#define EPS_STD   1e-5f
#define STAT_C    (TOWERS*HID)             /* 128 stats per node per array */

typedef __attribute__((ext_vector_type(16))) _Float16 v16h;
typedef __attribute__((ext_vector_type(8)))  float    v8f;

union Frag {
  uint4 q[2];
  v16h  v;
};

__device__ __forceinline__ unsigned packh2(float a, float b) {
  union { _Float16 h; unsigned short u; } ca, cb;
  ca.h = (_Float16)a; cb.h = (_Float16)b;
  return ((unsigned)cb.u << 16) | (unsigned)ca.u;
}

// ---------------------------------------------------------------------------
// h = x @ embW + embb ; also write f16 shadow. One wave per node (lane = out col).
__global__ void k_embed(const float* __restrict__ x, const float* __restrict__ W,
                        const float* __restrict__ b, float* __restrict__ h,
                        _Float16* __restrict__ hh) {
  int tid = blockIdx.x * blockDim.x + threadIdx.x;
  unsigned n = (unsigned)tid >> 5, j = tid & 31;
  if (n >= N_NODES) return;
  const float* xr = x + n * NODE_F;
  float acc = b[j];
  #pragma unroll
  for (int i = 0; i < NODE_F; i++) acc = fmaf(xr[i], W[i * HID + j], acc);
  h[n * HID + j] = acc;
  hh[n * HID + j] = (_Float16)acc;
}

__global__ void k_zero_deg(float* __restrict__ deg) {
  int i = blockIdx.x * blockDim.x + threadIdx.x;
  if (i < N_NODES) deg[i] = 0.0f;
}
__global__ void k_deg(const int* __restrict__ dst, float* __restrict__ deg) {
  int e = blockIdx.x * blockDim.x + threadIdx.x;
  if (e < N_EDGES) atomicAdd(&deg[(unsigned)dst[e]], 1.0f);
}

// ---------------------------------------------------------------------------
// Pre-swizzle preW (f32 -> f16) into WMMA B-fragment layout.
// Layout: [layer][tower][ntile][khalf][lane 0..31][dword j 0..7]
// Lane L holds column n = L&15; k-group = L>>4:
//   j<4  -> k = kg*8 + 2j,  j>=4 -> k = 16 + kg*8 + 2(j-4);  +32 for khalf=1.
__global__ void k_prep(const float* __restrict__ preW, unsigned* __restrict__ Bfrag) {
  int idx = blockIdx.x * blockDim.x + threadIdx.x;
  if (idx >= 2 * TOWERS * 2 * 2 * 32 * 8) return;
  int j    = idx & 7;
  int lane = (idx >> 3) & 31;
  int kh   = (idx >> 8) & 1;
  int nt   = (idx >> 9) & 1;
  int t    = (idx >> 10) & 3;
  int l    = (idx >> 12) & 1;
  int o  = nt * 16 + (lane & 15);
  int kg = lane >> 4;
  int k0 = (j < 4) ? (kg * 8 + 2 * j) : (16 + kg * 8 + 2 * (j - 4));
  k0 += kh * 32;
  const float* Wt = preW + (unsigned)(l * TOWERS + t) * (2 * HID) * HID;
  Bfrag[idx] = packh2(Wt[k0 * HID + o], Wt[(k0 + 1) * HID + o]);
}

// ---------------------------------------------------------------------------
__global__ void k_init_stats(float* __restrict__ sum, float* __restrict__ sumsq,
                             float* __restrict__ mn, float* __restrict__ mx) {
  unsigned i = blockIdx.x * blockDim.x + threadIdx.x;
  if (i >= (unsigned)N_NODES * STAT_C) return;
  sum[i]   = 0.0f;
  sumsq[i] = 0.0f;
  mn[i]    =  __builtin_inff();
  mx[i]    = -__builtin_inff();
}

// ---------------------------------------------------------------------------
// Edge kernel: one wave computes msg for 16 edges via WMMA and scatters stats.
//   A(k=0..31)  = h_half[dst]   (x_i half of cat)
//   A(k=32..63) = h_half[src]   (x_j half)
//   B fragments staged in LDS; bias pre-loaded into the WMMA C operand.
//   All stat offsets are unsigned 32-bit so atomics use saddr+voffset (GVS) form.
__global__ void __launch_bounds__(256) k_edge(
    const _Float16* __restrict__ hh,
    const int* __restrict__ srcA, const int* __restrict__ dstA,
    const unsigned* __restrict__ BfragG, const float* __restrict__ preb,
    float* __restrict__ sum, float* __restrict__ sumsq,
    float* __restrict__ mn, float* __restrict__ mx, int layer) {
  __shared__ __align__(16) unsigned sB[4096];   // 16 KB: this layer's B fragments
  const unsigned* bg = BfragG + layer * 4096;
  for (int i = threadIdx.x; i < 4096; i += blockDim.x) sB[i] = bg[i];
  __syncthreads();

  int lane  = threadIdx.x & 31;
  int wave  = blockIdx.x * (blockDim.x >> 5) + (threadIdx.x >> 5);
  int eBase = wave * 16;
  if (eBase >= N_EDGES) return;                 // wave-uniform: EXEC stays all-1 for WMMA

  int kg   = lane >> 4;
  int erow = eBase + (lane & 15);
  unsigned dr = (unsigned)dstA[erow];
  unsigned sr = (unsigned)srcA[erow];

  // A fragments: lane L covers edge-row (L&15); kg selects the 16-byte half-chunks.
  Frag Ad, As;
  const uint4* pd = (const uint4*)(hh + dr * HID);
  const uint4* ps = (const uint4*)(hh + sr * HID);
  Ad.q[0] = pd[kg];     Ad.q[1] = pd[2 + kg];
  As.q[0] = ps[kg];     As.q[1] = ps[2 + kg];

  // destination node for each of this lane's 8 accumulator rows (C layout: M = r + 8*kg)
  // 8 contiguous ints, 32-B aligned -> two b128 loads; pre-scale to element offsets.
  const int4* dp = (const int4*)(dstA + eBase + kg * 8);
  int4 d0 = dp[0], d1 = dp[1];
  unsigned dl[8];
  dl[0] = (unsigned)d0.x * STAT_C; dl[1] = (unsigned)d0.y * STAT_C;
  dl[2] = (unsigned)d0.z * STAT_C; dl[3] = (unsigned)d0.w * STAT_C;
  dl[4] = (unsigned)d1.x * STAT_C; dl[5] = (unsigned)d1.y * STAT_C;
  dl[6] = (unsigned)d1.z * STAT_C; dl[7] = (unsigned)d1.w * STAT_C;

  const float* pb = preb + layer * TOWERS * HID;

  #pragma unroll
  for (int t = 0; t < TOWERS; t++) {
    #pragma unroll
    for (int nt = 0; nt < 2; nt++) {
      Frag B0, B1;
      const uint4* b0 = (const uint4*)&sB[(((t * 2 + nt) * 2 + 0) << 8) + lane * 8];
      const uint4* b1 = (const uint4*)&sB[(((t * 2 + nt) * 2 + 1) << 8) + lane * 8];
      B0.q[0] = b0[0]; B0.q[1] = b0[1];
      B1.q[0] = b1[0]; B1.q[1] = b1[1];

      unsigned o   = nt * 16 + (lane & 15);
      float    bias = pb[t * HID + o];

      v8f c;                                    // C pre-loaded with bias (added once)
      #pragma unroll
      for (int i = 0; i < 8; i++) c[i] = bias;

      c = __builtin_amdgcn_wmma_f32_16x16x32_f16(false, Ad.v, false, B0.v,
                                                 (short)0, c, false, false);
      c = __builtin_amdgcn_wmma_f32_16x16x32_f16(false, As.v, false, B1.v,
                                                 (short)0, c, false, false);

      unsigned co = (unsigned)(t * HID) + o;
      #pragma unroll
      for (int r = 0; r < 8; r++) {
        float v = c[r];
        unsigned idx = dl[r] + co;              // 32-bit offset -> saddr atomic form
        atomicAdd(&sum[idx],   v);
        atomicAdd(&sumsq[idx], v * v);
        __hip_atomic_fetch_min(&mn[idx], v, __ATOMIC_RELAXED,
                               __HIP_MEMORY_SCOPE_AGENT);
        __hip_atomic_fetch_max(&mx[idx], v, __ATOMIC_RELAXED,
                               __HIP_MEMORY_SCOPE_AGENT);
      }
    }
  }
}

// ---------------------------------------------------------------------------
// Node update: stats -> scalers -> per-tower post-MLP -> tower-mixing linear.
__global__ void __launch_bounds__(256) k_node(
    const float* __restrict__ sum, const float* __restrict__ sumsq,
    const float* __restrict__ mn, const float* __restrict__ mx,
    const float* __restrict__ deg,
    const float* __restrict__ postW, const float* __restrict__ postb,
    const float* __restrict__ linW, const float* __restrict__ linb,
    float* __restrict__ h, _Float16* __restrict__ hh, int layer, int doRelu) {
  unsigned n = blockIdx.x * blockDim.x + threadIdx.x;
  if (n >= N_NODES) return;

  float d    = deg[n];
  float degc = fmaxf(d, 1.0f);
  float rdeg = 1.0f / degc;
  float amp  = logf(degc + 1.0f) * (1.0f / AVG_DEG_LOG);
  float iamp = 1.0f / amp;
  bool  has  = d > 0.0f;

  float hrow[HID];
  #pragma unroll
  for (int j = 0; j < HID; j++) hrow[j] = h[n * HID + j];

  float tout[TOWERS * F_OUT_T];
  const float* pwAll = postW + (unsigned)layer * TOWERS * 13 * HID * F_OUT_T;
  const float* pbias = postb + layer * TOWERS * F_OUT_T;

  for (int t = 0; t < TOWERS; t++) {
    const float* W = pwAll + (unsigned)t * 13 * HID * F_OUT_T;
    float acc[F_OUT_T];
    #pragma unroll
    for (int o = 0; o < F_OUT_T; o++) acc[o] = pbias[t * F_OUT_T + o];

    // section 0: node features
    for (int c = 0; c < HID; c++) {
      float hv = hrow[c];
      #pragma unroll
      for (int o = 0; o < F_OUT_T; o++) acc[o] = fmaf(hv, W[c * F_OUT_T + o], acc[o]);
    }

    unsigned base = n * STAT_C + t * HID;
    for (int c = 0; c < HID; c++) {
      float s1   = sum[base + c];
      float s2   = sumsq[base + c];
      float mean = s1 * rdeg;
      float var  = s2 * rdeg - mean * mean;
      var = var > 0.0f ? var : 0.0f;
      float stdv = sqrtf(var + EPS_STD);
      float mnv  = has ? mn[base + c] : 0.0f;
      float mxv  = has ? mx[base + c] : 0.0f;
      float stats[4] = {mean, mnv, mxv, stdv};
      #pragma unroll
      for (int s = 0; s < 4; s++) {
        int i1 = (1 + s) * HID + c;   // identity scaler block
        int i2 = (5 + s) * HID + c;   // * amp block
        int i3 = (9 + s) * HID + c;   // / amp block
        float st = stats[s];
        #pragma unroll
        for (int o = 0; o < F_OUT_T; o++) {
          float w = W[i1 * F_OUT_T + o]
                  + amp  * W[i2 * F_OUT_T + o]
                  + iamp * W[i3 * F_OUT_T + o];
          acc[o] = fmaf(st, w, acc[o]);
        }
      }
    }
    #pragma unroll
    for (int o = 0; o < F_OUT_T; o++) tout[t * F_OUT_T + o] = acc[o];
  }

  const float* LW = linW + layer * HID * HID;
  const float* Lb = linb + layer * HID;
  #pragma unroll
  for (int j = 0; j < HID; j++) {
    float v = Lb[j];
    #pragma unroll
    for (int k = 0; k < HID; k++) v = fmaf(tout[k], LW[k * HID + j], v);
    if (doRelu) v = fmaxf(v, 0.0f);
    h[n * HID + j]  = v;
    hh[n * HID + j] = (_Float16)v;
  }
}

// ---------------------------------------------------------------------------
__global__ void k_zero_g(float* __restrict__ g) {
  int i = blockIdx.x * blockDim.x + threadIdx.x;
  if (i < NUM_GRAPHS * HID) g[i] = 0.0f;
}
__global__ void k_pool(const float* __restrict__ h, const int* __restrict__ batch,
                       float* __restrict__ g) {
  unsigned tid = blockIdx.x * blockDim.x + threadIdx.x;
  if (tid >= (unsigned)N_NODES * HID) return;
  unsigned n = tid >> 5, j = tid & 31;
  atomicAdd(&g[(unsigned)batch[n] * HID + j], h[tid]);
}
__global__ void k_final(const float* __restrict__ g,
                        const float* __restrict__ W1, const float* __restrict__ b1,
                        const float* __restrict__ W2, const float* __restrict__ b2,
                        float* __restrict__ out) {
  int tid = threadIdx.x;
  if (tid >= NUM_GRAPHS * OUT_F) return;
  int gi = tid >> 3, o = tid & 7;
  float acc = b2[o];
  for (int k = 0; k < HID / 2; k++) {
    float hv = b1[k];
    #pragma unroll
    for (int j = 0; j < HID; j++) hv = fmaf(g[gi * HID + j], W1[j * (HID / 2) + k], hv);
    hv = fmaxf(hv, 0.0f);
    acc = fmaf(hv, W2[k * OUT_F + o], acc);
  }
  out[tid] = acc;
}

// ---------------------------------------------------------------------------
extern "C" void kernel_launch(void* const* d_in, const int* in_sizes, int n_in,
                              void* d_out, int out_size, void* d_ws, size_t ws_size,
                              hipStream_t stream) {
  const float* x     = (const float*)d_in[0];
  const int*   ei    = (const int*)d_in[1];
  const int*   batch = (const int*)d_in[2];
  const float* embW  = (const float*)d_in[3];
  const float* embb  = (const float*)d_in[4];
  const float* preW  = (const float*)d_in[5];
  const float* preb  = (const float*)d_in[6];
  const float* postW = (const float*)d_in[7];
  const float* postb = (const float*)d_in[8];
  const float* linW  = (const float*)d_in[9];
  const float* linb  = (const float*)d_in[10];
  const float* mlpW1 = (const float*)d_in[11];
  const float* mlpb1 = (const float*)d_in[12];
  const float* mlpW2 = (const float*)d_in[13];
  const float* mlpb2 = (const float*)d_in[14];
  float* out = (float*)d_out;

  const int* srcA = ei;
  const int* dstA = ei + N_EDGES;

  // workspace carve-up (256-B aligned slices)
  char* ws = (char*)d_ws;
  size_t off = 0;
  auto alloc = [&](size_t bytes) -> char* {
    char* p = ws + off;
    off += (bytes + 255) & ~(size_t)255;
    return p;
  };
  float*     h     = (float*)alloc((size_t)N_NODES * HID * 4);
  _Float16*  hh    = (_Float16*)alloc((size_t)N_NODES * HID * 2);
  float*     sum   = (float*)alloc((size_t)N_NODES * STAT_C * 4);
  float*     sumsq = (float*)alloc((size_t)N_NODES * STAT_C * 4);
  float*     mn    = (float*)alloc((size_t)N_NODES * STAT_C * 4);
  float*     mx    = (float*)alloc((size_t)N_NODES * STAT_C * 4);
  float*     deg   = (float*)alloc((size_t)N_NODES * 4);
  unsigned*  Bfrag = (unsigned*)alloc(8192 * 4);
  float*     g     = (float*)alloc((size_t)NUM_GRAPHS * HID * 4);

  dim3 blk(256);
  k_embed<<<(N_NODES * HID + 255) / 256, blk, 0, stream>>>(x, embW, embb, h, hh);
  k_zero_deg<<<(N_NODES + 255) / 256, blk, 0, stream>>>(deg);
  k_deg<<<(N_EDGES + 255) / 256, blk, 0, stream>>>(dstA, deg);
  k_prep<<<(8192 + 255) / 256, blk, 0, stream>>>(preW, Bfrag);

  const size_t STATN = (size_t)N_NODES * STAT_C;
  const int nWaves = N_EDGES / 16;          // 800000 / 16 = 50000 (exact)
  for (int l = 0; l < 2; l++) {
    for (int pass = 0; pass < 2; pass++) {  // same conv twice (reference bug), relu on 2nd
      k_init_stats<<<(int)((STATN + 255) / 256), blk, 0, stream>>>(sum, sumsq, mn, mx);
      k_edge<<<(nWaves + 7) / 8, blk, 0, stream>>>(hh, srcA, dstA, Bfrag, preb,
                                                   sum, sumsq, mn, mx, l);
      k_node<<<(N_NODES + 255) / 256, blk, 0, stream>>>(sum, sumsq, mn, mx, deg,
                                                        postW, postb, linW, linb,
                                                        h, hh, l, pass);
    }
  }

  k_zero_g<<<(NUM_GRAPHS * HID + 255) / 256, blk, 0, stream>>>(g);
  k_pool<<<(N_NODES * HID + 255) / 256, blk, 0, stream>>>(h, batch, g);
  k_final<<<1, 512, 0, stream>>>(g, mlpW1, mlpb1, mlpW2, mlpb2, out);
}
